// CrossAttentionNoGate_70239895159356
// MI455X (gfx1250) — compile-verified, matrix operand order checked by reading, not run on hardware
//
#include <hip/hip_runtime.h>
#include <hip/hip_bf16.h>

// ---------------------------------------------------------------------------
// CrossAttentionNoGate for MI455X (gfx1250, wave32, WMMA + TDM).
// B=4, Q=KV=2048, C_Q=C_KV=256, H=8, C_HID=32, d=256.
// Bias tensor (128 MB) dominates bandwidth -> bias-bound (~7us floor).
// GEMMs via v_wmma_f32_16x16x32_bf16; K/V^T tiles DMA'd to LDS by the
// Tensor Data Mover (tensor_load_to_lds, double-buffered, TENSORcnt waits).
// Workspace: qp 4MB | kp 4MB | vtp 4MB | ao 8MB  = 20MB.
// ---------------------------------------------------------------------------

#define BATCH 4
#define SEQ   2048
#define HEADS 8
#define CHID  32
#define DMODEL 256

typedef __attribute__((ext_vector_type(16))) __bf16 v16bf;
typedef __attribute__((ext_vector_type(8)))  float  v8f;
typedef __attribute__((ext_vector_type(4)))  unsigned int u32x4;
typedef __attribute__((ext_vector_type(8)))  unsigned int u32x8;

union Frag {
    unsigned int u[8];
    v16bf v;
};

__device__ __forceinline__ unsigned short f2bf(float f) {
    union { __bf16 h; unsigned short s; } u;
    u.h = (__bf16)f;               // native f32->bf16 (RNE) convert
    return u.s;
}

__device__ __forceinline__ unsigned int pack2(float lo, float hi) {
    union { __bf16 h[2]; unsigned int u; } p;
    p.h[0] = (__bf16)lo;
    p.h[1] = (__bf16)hi;
    return p.u;
}

__device__ __forceinline__ v8f wmma_bf16(const Frag& a, const Frag& b, v8f c) {
    // (neg_a, A, neg_b, B, c_mod, C, reuse_a, reuse_b)
    return __builtin_amdgcn_wmma_f32_16x16x32_bf16(false, a.v, false, b.v,
                                                   (short)0, c, false, false);
}

// Tensor Data Mover: 2D (or 1D when tile1==0) tile of 2-byte elements,
// global -> LDS.  D# groups per CDNA5 ISA.
// Groups 2/3 omitted (<=2D tensor).  Tensor dims set huge (no OOB clip).
__device__ __forceinline__ void tdm_load_tile(unsigned ldsAddr, const void* gp,
                                              unsigned tile0, unsigned tile1,
                                              unsigned long long stride0) {
    const unsigned long long ga = (unsigned long long)(uintptr_t)gp;
    u32x4 g0;
    g0[0] = 1u;                                    // count=1, user descriptor
    g0[1] = ldsAddr;                               // lds_addr (bytes)
    g0[2] = (unsigned)ga;                          // global_addr[31:0]
    g0[3] = (unsigned)((ga >> 32) & 0x01FFFFFFu)   // global_addr[56:32]
            | (2u << 30);                          // type=2 ("image")
    u32x8 g1;
    g1[0] = 1u << 16;                              // data_size=1 (2 bytes)
    g1[1] = 0u;                                    // tensor_dim0[15:0]=0
    g1[2] = 0x4000u;                               // tensor_dim0=0x40000000
    g1[3] = 0x4000u | (tile0 << 16);               // tensor_dim1 hi | tile_dim0
    g1[4] = tile1;                                 // tile_dim1 (0 => 1D)
    g1[5] = (unsigned)stride0;                     // tensor_dim0_stride lo
    g1[6] = (unsigned)(stride0 >> 32);             // tensor_dim0_stride hi
    g1[7] = 0u;
    asm volatile("tensor_load_to_lds %0, %1" :: "s"(g0), "s"(g1) : "memory");
}

// 16-bit A/B fragment: dword j of lane (half h2) holds K pair starting at
//   k = (j>=4 ? 16 : 0) + h2*8 + (j&3)*2        (element index)
//   d = (j>=4 ? 8  : 0) + h2*4 + (j&3)          (dword index)

// ---------------------------------------------------------------------------
// Kernel 1: QKV projection.  grid(512, 8, 3), block(32).
//   z=0: q = (input_q @ w_q) * 1/sqrt(32) -> qp [B,H,Q,32] bf16
//   z=1: k = input_kv @ w_k              -> kp [B,H,KV,32] bf16
//   z=2: v = input_kv @ w_v (transposed) -> vtp [B,H,32,KV] bf16
// ---------------------------------------------------------------------------
__global__ void __launch_bounds__(32)
proj_qkv_kernel(const float* __restrict__ xq, const float* __restrict__ xkv,
                const float* __restrict__ wq, const float* __restrict__ wk,
                const float* __restrict__ wv,
                unsigned short* __restrict__ qp,
                unsigned short* __restrict__ kp,
                unsigned short* __restrict__ vtp) {
    const int lane = threadIdx.x & 31;
    const int ln = lane & 15, h2 = lane >> 4;
    const int mode = blockIdx.z;
    const int tile = blockIdx.x;     // 16-row tile of the 8192 flattened tokens
    const int head = blockIdx.y;     // 32-wide N tile == one head

    const float* X = (mode == 0) ? xq : xkv;
    const float* W = (mode == 0) ? wq : ((mode == 1) ? wk : wv);

    const float* xrow = X + (size_t)(tile * 16 + ln) * DMODEL;
    const int n0 = head * 32 + ln;

    v8f c0 = {}, c1 = {};
    for (int k0 = 0; k0 < DMODEL; k0 += 32) {
        Frag a, b0, b1;
#pragma unroll
        for (int j = 0; j < 8; ++j) {
            const int k = k0 + ((j & 4) << 2) + h2 * 8 + ((j & 3) << 1);
            a.u[j]  = pack2(xrow[k], xrow[k + 1]);
            b0.u[j] = pack2(W[(size_t)k * DMODEL + n0],
                            W[(size_t)(k + 1) * DMODEL + n0]);
            b1.u[j] = pack2(W[(size_t)k * DMODEL + n0 + 16],
                            W[(size_t)(k + 1) * DMODEL + n0 + 16]);
        }
        c0 = wmma_bf16(a, b0, c0);
        c1 = wmma_bf16(a, b1, c1);
    }

    const float qscale = 0.17677669529663687f;  // 1/sqrt(C_HID)
#pragma unroll
    for (int r = 0; r < 8; ++r) {
        const int row  = tile * 16 + r + 8 * h2;   // flattened token id
        const int bidx = row >> 11;                // / 2048
        const int pos  = row & 2047;
        if (mode == 0) {
            const size_t base = (((size_t)bidx * HEADS + head) * SEQ + pos) * CHID;
            qp[base + ln]      = f2bf(c0[r] * qscale);
            qp[base + 16 + ln] = f2bf(c1[r] * qscale);
        } else if (mode == 1) {
            const size_t base = (((size_t)bidx * HEADS + head) * SEQ + pos) * CHID;
            kp[base + ln]      = f2bf(c0[r]);
            kp[base + 16 + ln] = f2bf(c1[r]);
        } else {
            const size_t base = (((size_t)bidx * HEADS + head) * CHID) * (size_t)SEQ + pos;
            vtp[base + (size_t)ln * SEQ]        = f2bf(c0[r]);
            vtp[base + (size_t)(16 + ln) * SEQ] = f2bf(c1[r]);
        }
    }
}

// ---------------------------------------------------------------------------
// Kernel 2: flash attention.  grid(Q/128=16, H=8, B=4), block(256) = 8 waves.
// Wave w owns q rows [qbase, qbase+16).  KV streamed in chunks of 32;
// K / V^T chunk tiles are DMA'd into LDS by the TDM (wave 0), double-buffered.
// ---------------------------------------------------------------------------
__global__ void __launch_bounds__(256)
attn_kernel(const unsigned short* __restrict__ qp,
            const unsigned short* __restrict__ kp,
            const unsigned short* __restrict__ vtp,
            const float* __restrict__ mask,
            const float* __restrict__ bias,
            float* __restrict__ ao) {
    __shared__ unsigned short kbuf[2][32][32];   // [buf][kv_local][c]   4 KB
    __shared__ unsigned short vbuf[2][32][32];   // [buf][c][kv_local]   4 KB
    __shared__ unsigned short plds[8][16][32];   // per-wave P staging   8 KB

    const int tid  = threadIdx.x;
    const int w    = tid >> 5;
    const int lane = tid & 31;
    const int ln   = lane & 15, h2 = lane >> 4;
    const int b = blockIdx.z, hy = blockIdx.y;
    const int bh = b * HEADS + hy;
    const int qbase = blockIdx.x * 128 + w * 16;

    const unsigned short* kbase = kp + (size_t)bh * SEQ * CHID;
    const unsigned short* vbase = vtp + (size_t)bh * CHID * SEQ;

    // ---- load Q A-fragment (persistent) ----
    Frag qa;
    {
        const unsigned int* qrow =
            (const unsigned int*)(qp + ((size_t)bh * SEQ + qbase + ln) * CHID);
#pragma unroll
        for (int j = 0; j < 8; ++j)
            qa.u[j] = qrow[((j & 4) << 1) + h2 * 4 + (j & 3)];
    }

    v8f o0 = {}, o1 = {};
    float mrow[8], lrow[8];
#pragma unroll
    for (int r = 0; r < 8; ++r) { mrow[r] = -1e30f; lrow[r] = 0.f; }

    // bias row base for this wave's half: row(r) = qbase + r + 8*h2
    const float* biasq = bias + ((size_t)hy * SEQ + qbase + 8 * h2) * (size_t)SEQ;
    const float* maskb = mask + (size_t)b * SEQ;

    // ---- TDM preload of chunk 0 (wave 0 only; EXEC-independent DMA) ----
    if (w == 0) {
        tdm_load_tile((unsigned)(uintptr_t)&kbuf[0][0][0], kbase,
                      1024u, 0u, 1024ull);            // contiguous 2 KB
        tdm_load_tile((unsigned)(uintptr_t)&vbuf[0][0][0], vbase,
                      32u, 32u, (unsigned long long)SEQ);  // 32x32, stride SEQ
    }

    const int nch = SEQ / 32;
    for (int ch = 0; ch < nch; ++ch) {
        const int kv0 = ch * 32;
        const int buf = ch & 1;

        if (w == 0) {
            if (ch + 1 < nch) {
                const int kv1 = kv0 + 32;
                tdm_load_tile((unsigned)(uintptr_t)&kbuf[buf ^ 1][0][0],
                              kbase + (size_t)kv1 * CHID, 1024u, 0u, 1024ull);
                tdm_load_tile((unsigned)(uintptr_t)&vbuf[buf ^ 1][0][0],
                              vbase + kv1, 32u, 32u, (unsigned long long)SEQ);
                __builtin_amdgcn_s_wait_tensorcnt(2);  // chunk ch done
            } else {
                __builtin_amdgcn_s_wait_tensorcnt(0);
            }
        }
        __syncthreads();   // chunk ch visible to all 8 waves

        // ---- K B-fragments from LDS (columns = kv positions) ----
        Frag kb0, kb1;
        {
            const unsigned int* kr0 = (const unsigned int*)&kbuf[buf][ln][0];
            const unsigned int* kr1 = (const unsigned int*)&kbuf[buf][16 + ln][0];
#pragma unroll
            for (int j = 0; j < 8; ++j) {
                const int d = ((j & 4) << 1) + h2 * 4 + (j & 3);
                kb0.u[j] = kr0[d];
                kb1.u[j] = kr1[d];
            }
        }
        v8f zero = {};
        v8f s0 = wmma_bf16(qa, kb0, zero);
        v8f s1 = wmma_bf16(qa, kb1, zero);

        const float mt0 = (maskb[kv0 + ln] - 1.f) * 1e9f;
        const float mt1 = (maskb[kv0 + 16 + ln] - 1.f) * 1e9f;

        // prefetch next bias chunk into cache hierarchy (global_prefetch_b8)
        if (kv0 + 32 < SEQ)
            __builtin_prefetch(biasq + (size_t)kv0 + 32 + ln, 0, 3);

#pragma unroll
        for (int r = 0; r < 8; ++r) {
            const float* brow = biasq + (size_t)r * SEQ + kv0;
            float a0 = s0[r] + brow[ln] + mt0;
            float a1 = s1[r] + brow[16 + ln] + mt1;

            // row max over the 32-wide chunk: reduce across 16-lane half group
            float t = fmaxf(a0, a1);
            t = fmaxf(t, __shfl_xor(t, 1, 32));
            t = fmaxf(t, __shfl_xor(t, 2, 32));
            t = fmaxf(t, __shfl_xor(t, 4, 32));
            t = fmaxf(t, __shfl_xor(t, 8, 32));

            const float mnew  = fmaxf(mrow[r], t);
            const float alpha = __expf(mrow[r] - mnew);
            const float p0 = __expf(a0 - mnew);
            const float p1 = __expf(a1 - mnew);

            float ps = p0 + p1;
            ps += __shfl_xor(ps, 1, 32);
            ps += __shfl_xor(ps, 2, 32);
            ps += __shfl_xor(ps, 4, 32);
            ps += __shfl_xor(ps, 8, 32);

            lrow[r] = lrow[r] * alpha + ps;
            mrow[r] = mnew;
            o0[r] *= alpha;
            o1[r] *= alpha;

            plds[w][r + 8 * h2][ln]      = f2bf(p0);
            plds[w][r + 8 * h2][16 + ln] = f2bf(p1);
        }

        // same-wave LDS ops are in-order; block compiler reordering only
        asm volatile("" ::: "memory");

        // ---- P A-fragment from LDS ----
        Frag pa;
        {
            const unsigned int* pr = (const unsigned int*)&plds[w][ln][0];
#pragma unroll
            for (int j = 0; j < 8; ++j)
                pa.u[j] = pr[((j & 4) << 1) + h2 * 4 + (j & 3)];
        }

        // ---- V^T B-fragments from LDS (columns = output channel c) ----
        Frag vb0, vb1;
        {
            const unsigned int* vr0 = (const unsigned int*)&vbuf[buf][ln][0];
            const unsigned int* vr1 = (const unsigned int*)&vbuf[buf][16 + ln][0];
#pragma unroll
            for (int j = 0; j < 8; ++j) {
                const int d = ((j & 4) << 1) + h2 * 4 + (j & 3);
                vb0.u[j] = vr0[d];
                vb1.u[j] = vr1[d];
            }
        }
        o0 = wmma_bf16(pa, vb0, o0);
        o1 = wmma_bf16(pa, vb1, o1);

        __syncthreads();   // all waves done with buf before TDM overwrites it
    }

    // ---- normalize + store attention output [B, Q, H*32] f32 ----
#pragma unroll
    for (int r = 0; r < 8; ++r) {
        const int qr = qbase + r + 8 * h2;
        const float inv = 1.f / lrow[r];
        const size_t base = ((size_t)b * SEQ + qr) * DMODEL + hy * CHID;
        ao[base + ln]      = o0[r] * inv;
        ao[base + 16 + ln] = o1[r] * inv;
    }
}

// ---------------------------------------------------------------------------
// Kernel 3: output projection  out = ao @ w_o + b_o.  grid(512, 8), block(32).
// ---------------------------------------------------------------------------
__global__ void __launch_bounds__(32)
outproj_kernel(const float* __restrict__ X, const float* __restrict__ W,
               const float* __restrict__ bo, float* __restrict__ out) {
    const int lane = threadIdx.x & 31;
    const int ln = lane & 15, h2 = lane >> 4;
    const int tile = blockIdx.x, nb = blockIdx.y;

    const float* xrow = X + (size_t)(tile * 16 + ln) * DMODEL;
    const int n0 = nb * 32 + ln;

    v8f c0 = {}, c1 = {};
    for (int k0 = 0; k0 < DMODEL; k0 += 32) {
        Frag a, b0, b1;
#pragma unroll
        for (int j = 0; j < 8; ++j) {
            const int k = k0 + ((j & 4) << 2) + h2 * 8 + ((j & 3) << 1);
            a.u[j]  = pack2(xrow[k], xrow[k + 1]);
            b0.u[j] = pack2(W[(size_t)k * DMODEL + n0],
                            W[(size_t)(k + 1) * DMODEL + n0]);
            b1.u[j] = pack2(W[(size_t)k * DMODEL + n0 + 16],
                            W[(size_t)(k + 1) * DMODEL + n0 + 16]);
        }
        c0 = wmma_bf16(a, b0, c0);
        c1 = wmma_bf16(a, b1, c1);
    }

    const float bo0 = bo[n0], bo1 = bo[n0 + 16];
#pragma unroll
    for (int r = 0; r < 8; ++r) {
        const size_t row = (size_t)(tile * 16 + r + 8 * h2);
        out[row * DMODEL + n0]      = c0[r] + bo0;
        out[row * DMODEL + n0 + 16] = c1[r] + bo1;
    }
}

// ---------------------------------------------------------------------------
extern "C" void kernel_launch(void* const* d_in, const int* in_sizes, int n_in,
                              void* d_out, int out_size, void* d_ws, size_t ws_size,
                              hipStream_t stream) {
    const float* input_q  = (const float*)d_in[0];
    const float* input_kv = (const float*)d_in[1];
    const float* mask     = (const float*)d_in[2];
    const float* bias     = (const float*)d_in[3];
    const float* w_q      = (const float*)d_in[4];
    const float* w_k      = (const float*)d_in[5];
    const float* w_v      = (const float*)d_in[6];
    const float* w_o      = (const float*)d_in[7];
    const float* b_o      = (const float*)d_in[8];
    float* out = (float*)d_out;

    char* ws = (char*)d_ws;
    unsigned short* qp  = (unsigned short*)(ws);                         // 4 MB
    unsigned short* kp  = (unsigned short*)(ws + ((size_t)4 << 20));     // 4 MB
    unsigned short* vtp = (unsigned short*)(ws + ((size_t)8 << 20));     // 4 MB
    float*          ao  = (float*)         (ws + ((size_t)12 << 20));    // 8 MB

    // 1) QKV projections (bf16 WMMA)
    proj_qkv_kernel<<<dim3(512, 8, 3), dim3(32), 0, stream>>>(
        input_q, input_kv, w_q, w_k, w_v, qp, kp, vtp);

    // 2) Flash attention with bias + mask (bf16 WMMA, TDM-staged K/V)
    attn_kernel<<<dim3(16, 8, 4), dim3(256), 0, stream>>>(
        qp, kp, vtp, mask, bias, ao);

    // 3) Output projection + bias
    outproj_kernel<<<dim3(512, 8), dim3(32), 0, stream>>>(ao, w_o, b_o, out);
}
